// MultiHeadQuadraticStateSpaceAttention_59150289600734
// MI455X (gfx1250) — compile-verified, matrix operand order checked by reading, not run on hardware
//
#include <hip/hip_runtime.h>
#include <hip/hip_bf16.h>

// ---------------------------------------------------------------------------
// MultiHeadQuadraticStateSpaceAttention for MI455X (gfx1250, wave32, WMMA)
//
// bf16 WMMA (f32 accumulate) for all GEMMs; the dominant output projection
// streams its operands into LDS with the Tensor Data Mover (tensor_load_to_lds,
// TENSORcnt-tracked, double-buffered), using TDM LDS padding to keep a
// 144-byte row pitch for conflict-friendly fragment loads.
// ---------------------------------------------------------------------------

typedef __attribute__((ext_vector_type(16))) __bf16 v16bf;
typedef __attribute__((ext_vector_type(8)))  __bf16 v8bf;
typedef __attribute__((ext_vector_type(8)))  float  v8f;
typedef __attribute__((ext_vector_type(4)))  unsigned int v4u;
typedef __attribute__((ext_vector_type(8)))  int    v8i;
typedef __attribute__((ext_vector_type(4)))  int    v4i;

constexpr int Bc = 4, Tc = 4096, Ec = 1024, Hc = 16, Sc = 64, Dc = 64;
constexpr int ROWS = Bc * Tc;   // 16384 token rows
constexpr int LDK  = 72;        // LDS row pitch in bf16 (144 B, multiple of 16 B)

// ---- WMMA helpers ---------------------------------------------------------
// 16-bit A (16x32) fragment layout (ISA 7.12.2): lane L -> row = L%16,
// k-half = L/16; VGPR0..3 hold K = 8*khalf + 0..7, VGPR4..7 hold K+16.
// B (32x16) stored K-contiguous per column behaves identically, so one loader
// serves both A ([M][K] in LDS) and B ([N][K] in LDS).
static __device__ __forceinline__ v16bf frag_ld(const __bf16* base, int row0,
                                                int kbase, int lane) {
  const int row = row0 + (lane & 15);
  const int kh  = (lane >> 4) << 3;                 // 0 or 8
  const __bf16* p = base + row * LDK + kbase + kh;  // 16B aligned
  v8bf lo = *(const v8bf*)(p);                      // K .. K+7
  v8bf hi = *(const v8bf*)(p + 16);                 // K+16 .. K+23
  return __builtin_shufflevector(lo, hi, 0, 1, 2, 3, 4, 5, 6, 7,
                                 8, 9, 10, 11, 12, 13, 14, 15);
}

static __device__ __forceinline__ v8f wmma_bf16(v16bf a, v16bf b, v8f c) {
  return __builtin_amdgcn_wmma_f32_16x16x32_bf16(false, a, false, b,
                                                 (short)0, c, false, false);
}

// ---- TDM: DMA a 2-D bf16 tile (rows x 64, row stride Ec elems) to LDS ------
// D# packing per ISA ch.8: group0 = {count/type, lds_addr, global_addr},
// group1 = {data_size=2B, pad(128B rows -> +16B), tensor/tile dims, stride}.
// Padding: pad_interval code 4 (every 32 DWORDs = 64 bf16) + pad_amount
// code 3 (4 DWORDs = 8 bf16) reproduces the LDK=72 pitch.
static __device__ __forceinline__ void tdm_load_tile(unsigned int lds_addr,
                                                     const __bf16* gptr,
                                                     int rows) {
  const unsigned long long ga = (unsigned long long)(uintptr_t)gptr;
  const v4u g0 = {
      1u,                                                   // count=1, user D#
      lds_addr,                                             // LDS byte address
      (unsigned int)ga,                                     // global_addr lo
      (unsigned int)((ga >> 32) & 0x1FFFFFFu) | (2u << 30)  // ga hi | type=2
  };
  const v8i g1 = {
      (int)((1u << 16) | (1u << 20) | (4u << 22) | (3u << 25)),
      //     data_size=2B  pad_en      pad_intvl=32dw pad_amt=4dw
      (int)((unsigned)(Ec & 0xFFFF) << 16),          // tensor_dim0 lo16
      (int)((Ec >> 16) | ((unsigned)rows << 16)),    // tdim0 hi | tdim1 lo16
      (int)(64u << 16),                              // tdim1 hi=0 | tile_dim0=64
      (int)rows,                                     // tile_dim1 | tile_dim2=0
      (int)Ec,                                       // tensor_dim0_stride lo32
      0,                                             // stride0 hi | stride1 lo
      0                                              // stride1 hi
  };
  const v4i z4 = {0, 0, 0, 0};                       // 2-D: groups 2/3 unused
  const v8i z8 = {0, 0, 0, 0, 0, 0, 0, 0};
  // 6-arg variant (clang-23 / therock headers)
  __builtin_amdgcn_tensor_load_to_lds(g0, g1, z4, z4, z8, 0);
}

// ---------------------------------------------------------------------------
// Kernel 0: a_state[h,b,s] = state[h,b,:] . A_w[h,s,:] + A_b[h,s]   (tiny)
// ---------------------------------------------------------------------------
__global__ __launch_bounds__(64) void k_astate(const float* __restrict__ state,
                                               const float* __restrict__ Aw,
                                               const float* __restrict__ Ab,
                                               float* __restrict__ astate) {
  const int hb = blockIdx.x;          // h*B + b
  const int h  = hb >> 2;             // B == 4
  const int s  = threadIdx.x;         // 0..63
  const float* stp = state + (size_t)hb * Sc;
  const float* awp = Aw + ((size_t)h * Sc + s) * Sc;
  float acc = Ab[(size_t)h * Sc + s];
#pragma unroll 8
  for (int j = 0; j < Sc; ++j) acc += stp[j] * awp[j];
  astate[(size_t)hb * Sc + s] = acc;
}

// ---------------------------------------------------------------------------
// Kernel 0b: Wo f32 -> bf16 (so TDM can stream it; 2 MB, stays L2-resident)
// ---------------------------------------------------------------------------
__global__ __launch_bounds__(256) void k_cvtwo(const float* __restrict__ Wo,
                                               __bf16* __restrict__ Wob) {
  const size_t i = ((size_t)blockIdx.x * 256 + threadIdx.x) * 4;
  const float4 v = *(const float4*)(Wo + i);
  Wob[i + 0] = (__bf16)v.x;
  Wob[i + 1] = (__bf16)v.y;
  Wob[i + 2] = (__bf16)v.z;
  Wob[i + 3] = (__bf16)v.w;
}

// ---------------------------------------------------------------------------
// Kernel 1: per (128-row tile, head h)
//   ns    = x_h(128x64) @ sm[h]^T  + a_state[h,b,:]   -> f32 new_state out
//   out_h = ns(128x64)  @ C_w[h]^T + C_b[h]           -> bf16 scratch
// ---------------------------------------------------------------------------
__global__ __launch_bounds__(256) void k_head(const float* __restrict__ x,
                                              const float* __restrict__ sm,
                                              const float* __restrict__ Cw,
                                              const float* __restrict__ Cb,
                                              const float* __restrict__ astate,
                                              __bf16* __restrict__ outh,
                                              float* __restrict__ nstate) {
  __shared__ __align__(16) __bf16 xs[128 * LDK];   // x tile,   bf16
  __shared__ __align__(16) __bf16 wsm[64 * LDK];   // sm[h] / C_w[h]  ([N][K])
  __shared__ __align__(16) __bf16 nss[128 * LDK];  // new_state tile, bf16
  const int h   = blockIdx.y;
  const int bt0 = blockIdx.x * 128;
  const int b   = bt0 / Tc;          // T % 128 == 0 -> tile within one batch
  const int t0  = bt0 - b * Tc;
  const int tid = threadIdx.x, lane = tid & 31, wave = tid >> 5;

  // x tile: 128 x 64 f32 -> bf16 LDS
  for (int i = tid; i < 128 * Dc / 4; i += 256) {
    const int r = i >> 4, c = (i & 15) << 2;
    const float4 v = *(const float4*)(x + (size_t)(bt0 + r) * Ec + h * Dc + c);
    xs[r * LDK + c + 0] = (__bf16)v.x;
    xs[r * LDK + c + 1] = (__bf16)v.y;
    xs[r * LDK + c + 2] = (__bf16)v.z;
    xs[r * LDK + c + 3] = (__bf16)v.w;
  }
  // sm[h]: [S][D] f32 -> bf16 LDS  (N=s rows, K=d contiguous)
  for (int i = tid; i < Sc * Dc / 4; i += 256) {
    const int r = i >> 4, c = (i & 15) << 2;
    const float4 v = *(const float4*)(sm + ((size_t)h * Sc + r) * Dc + c);
    wsm[r * LDK + c + 0] = (__bf16)v.x;
    wsm[r * LDK + c + 1] = (__bf16)v.y;
    wsm[r * LDK + c + 2] = (__bf16)v.z;
    wsm[r * LDK + c + 3] = (__bf16)v.w;
  }
  __syncthreads();

  const int m0    = wave * 16;   // 8 waves -> 8 m-tiles of the 128 rows
  const int khalf = lane >> 4;

  // GEMM1: ns = x @ sm^T
  v8f acc[4] = {};
#pragma unroll
  for (int ks = 0; ks < 64; ks += 32) {
    const v16bf a = frag_ld(xs, m0, ks, lane);
#pragma unroll
    for (int n = 0; n < 4; ++n) {
      const v16bf bf = frag_ld(wsm, n * 16, ks, lane);
      acc[n] = wmma_bf16(a, bf, acc[n]);
    }
  }
  // epilogue: + a_state broadcast, emit f32 new_state ([H,B,T,S]) + bf16 LDS
#pragma unroll
  for (int n = 0; n < 4; ++n) {
    const int s = n * 16 + (lane & 15);
    const float as = astate[((size_t)h * Bc + b) * Sc + s];
#pragma unroll
    for (int r = 0; r < 8; ++r) {
      const int row = m0 + r + 8 * khalf;
      const float v = acc[n][r] + as;
      nstate[(((size_t)h * Bc + b) * Tc + t0 + row) * Sc + s] = v;
      nss[row * LDK + s] = (__bf16)v;
    }
  }
  __syncthreads();

  // C_w[h]: [D][S] f32 -> bf16 LDS (reuse weight buffer)
  for (int i = tid; i < Dc * Sc / 4; i += 256) {
    const int r = i >> 4, c = (i & 15) << 2;
    const float4 v = *(const float4*)(Cw + ((size_t)h * Dc + r) * Sc + c);
    wsm[r * LDK + c + 0] = (__bf16)v.x;
    wsm[r * LDK + c + 1] = (__bf16)v.y;
    wsm[r * LDK + c + 2] = (__bf16)v.z;
    wsm[r * LDK + c + 3] = (__bf16)v.w;
  }
  __syncthreads();

  // GEMM2: out_h = ns @ C_w^T + C_b   -> bf16 scratch [ROWS][E]
  v8f acc2[4] = {};
#pragma unroll
  for (int ks = 0; ks < 64; ks += 32) {
    const v16bf a = frag_ld(nss, m0, ks, lane);
#pragma unroll
    for (int n = 0; n < 4; ++n) {
      const v16bf bf = frag_ld(wsm, n * 16, ks, lane);
      acc2[n] = wmma_bf16(a, bf, acc2[n]);
    }
  }
#pragma unroll
  for (int n = 0; n < 4; ++n) {
    const int d = n * 16 + (lane & 15);
    const float cb = Cb[(size_t)h * Dc + d];
#pragma unroll
    for (int r = 0; r < 8; ++r) {
      const int row = m0 + r + 8 * khalf;
      outh[(size_t)(bt0 + row) * Ec + h * Dc + d] = (__bf16)(acc2[n][r] + cb);
    }
  }
}

// ---------------------------------------------------------------------------
// Kernel 2: out[ROWS,E] = out_h(bf16) @ Wo^T + bo   (34.4 GFLOP)
// 128x64 tile per block; TDM double-buffered LDS staging; 8 waves, each wave
// one 16-row strip x 4 n-tiles of v_wmma_f32_16x16x32_bf16.
// ---------------------------------------------------------------------------
__global__ __launch_bounds__(256) void k_out(const __bf16* __restrict__ A,
                                             const __bf16* __restrict__ Wob,
                                             const float* __restrict__ bo,
                                             float* __restrict__ out) {
  __shared__ __align__(16) __bf16 As[2][128 * LDK];  // 2 x 18 KB
  __shared__ __align__(16) __bf16 Bs[2][64 * LDK];   // 2 x 9 KB
  const int m0g = blockIdx.x * 128;
  const int n0g = blockIdx.y * 64;
  const int tid = threadIdx.x, lane = tid & 31, wave = tid >> 5;

  const __bf16* Ag = A   + (size_t)m0g * Ec;  // 128 rows, K-major
  const __bf16* Bg = Wob + (size_t)n0g * Ec;  // 64 rows (out cols), K-major

  // stage 0 prologue: wave 0 issues both DMAs (EXEC-independent, per-wave op)
  if (tid < 32) {
    tdm_load_tile((unsigned int)(uintptr_t)(void*)&As[0][0], Ag, 128);
    tdm_load_tile((unsigned int)(uintptr_t)(void*)&Bs[0][0], Bg, 64);
  }

  v8f acc[4] = {};
  int buf = 0;
  for (int kc = 0; kc < Ec; kc += 64) {
    if (kc + 64 < Ec) {
      if (tid < 32) {
        // prefetch next chunk into the other buffer, then retire current pair
        tdm_load_tile((unsigned int)(uintptr_t)(void*)&As[buf ^ 1][0],
                      Ag + kc + 64, 128);
        tdm_load_tile((unsigned int)(uintptr_t)(void*)&Bs[buf ^ 1][0],
                      Bg + kc + 64, 64);
        __builtin_amdgcn_s_wait_tensorcnt(2);
      }
    } else {
      if (tid < 32) __builtin_amdgcn_s_wait_tensorcnt(0);
    }
    __syncthreads();  // publish TDM-filled LDS to all waves

    const __bf16* Abuf = &As[buf][0];
    const __bf16* Bbuf = &Bs[buf][0];
    const int m0 = wave * 16;
#pragma unroll
    for (int ks = 0; ks < 64; ks += 32) {
      const v16bf a = frag_ld(Abuf, m0, ks, lane);
#pragma unroll
      for (int n = 0; n < 4; ++n) {
        const v16bf bf = frag_ld(Bbuf, n * 16, ks, lane);
        acc[n] = wmma_bf16(a, bf, acc[n]);
      }
    }
    __syncthreads();  // all waves done reading before buffer is re-DMA'd
    buf ^= 1;
  }

  const int khalf = lane >> 4;
  const int m0 = wave * 16;
#pragma unroll
  for (int n = 0; n < 4; ++n) {
    const int col = n0g + n * 16 + (lane & 15);
    const float bias = bo[col];
#pragma unroll
    for (int r = 0; r < 8; ++r) {
      out[(size_t)(m0g + m0 + r + 8 * khalf) * Ec + col] = acc[n][r] + bias;
    }
  }
}

// ---------------------------------------------------------------------------
extern "C" void kernel_launch(void* const* d_in, const int* in_sizes, int n_in,
                              void* d_out, int out_size, void* d_ws,
                              size_t ws_size, hipStream_t stream) {
  (void)in_sizes; (void)n_in; (void)out_size; (void)ws_size;
  const float* x  = (const float*)d_in[0];
  const float* st = (const float*)d_in[1];
  const float* Aw = (const float*)d_in[2];
  const float* Ab = (const float*)d_in[3];
  const float* Cw = (const float*)d_in[4];
  const float* Cb = (const float*)d_in[5];
  const float* sm = (const float*)d_in[6];
  const float* Wo = (const float*)d_in[7];
  const float* bo = (const float*)d_in[8];

  float* out    = (float*)d_out;                 // [B,T,E] f32
  float* nstate = out + (size_t)ROWS * Ec;       // [H,B,T,S] f32

  // workspace: a_state f32 (16 KB) | out_h bf16 (32 MB) | Wo bf16 (2 MB)
  char* wsb = (char*)d_ws;
  float*  astate = (float*)wsb;
  __bf16* outh   = (__bf16*)(wsb + (size_t)Hc * Bc * Sc * sizeof(float));
  __bf16* wob    = outh + (size_t)ROWS * Ec;

  k_astate<<<dim3(Hc * Bc), dim3(64), 0, stream>>>(st, Aw, Ab, astate);
  k_cvtwo<<<dim3((Ec * Ec) / (256 * 4)), dim3(256), 0, stream>>>(Wo, wob);
  k_head<<<dim3(ROWS / 128, Hc), dim3(256), 0, stream>>>(x, sm, Cw, Cb, astate,
                                                         outh, nstate);
  k_out<<<dim3(ROWS / 128, Ec / 64), dim3(256), 0, stream>>>(outh, wob, bo, out);
}
